// SR_LSTM_74242804678677
// MI455X (gfx1250) — matrix operand order repaired and checked by reading, not compiled, not run on hardware
//
#include <hip/hip_runtime.h>
#include <math.h>
#include <stdint.h>

#define T_STEPS 20
#define NN 192
#define DH 64
#define DEMB 32
#define DREL 32

typedef __attribute__((ext_vector_type(16))) _Float16 v16h;
typedef __attribute__((ext_vector_type(8)))  float    v8f;

__device__ __forceinline__ float sigf(float x){ return 1.0f/(1.0f+__expf(-x)); }

// Async global->LDS copy of one dword per lane (CDNA5 GLOBAL_LOAD_ASYNC_TO_LDS_B32).
// ldsaddr = wave-relative LDS byte address (low 32 bits of generic LDS pointer).
__device__ __forceinline__ void async_ld_b32(uint32_t ldsaddr, const float* gaddr){
  asm volatile("global_load_async_to_lds_b32 %0, %1, off"
               :: "v"(ldsaddr), "v"(gaddr) : "memory");
}
__device__ __forceinline__ void wait_async(){
  asm volatile("s_wait_asynccnt 0x0" ::: "memory");
}

// ---------------------------------------------------------------- init
__global__ void init_k(float* out, float* vacc, int n){
  int idx = blockIdx.x*blockDim.x+threadIdx.x;
  if (idx < n) out[idx] = 0.f;
  if (idx < 3) vacc[idx] = 0.f;
}

// ---------------------------------------------------------------- LSTM cell (one block per node, 64 threads)
__global__ void lstm_k(const float* __restrict__ norm,
                       const float* __restrict__ w_in, const float* __restrict__ b_in,
                       const float* __restrict__ w_ih, const float* __restrict__ w_hh,
                       const float* __restrict__ b_ih, const float* __restrict__ b_hh,
                       const float* __restrict__ hcar, const float* __restrict__ ccar,
                       float* hA, float* cA, float* oG, float* scal, int t){
  __shared__ float xb[DEMB];
  __shared__ float hb[DH];
  int i = blockIdx.x, h = threadIdx.x;
  if (h < DEMB){
    float n0 = norm[(t*NN+i)*2+0], n1 = norm[(t*NN+i)*2+1];
    float v = n0*w_in[h*2+0] + n1*w_in[h*2+1] + b_in[h];
    xb[h] = v > 0.f ? v : 0.f;
  }
  hb[h] = hcar[i*DH+h];
  if (i==0 && h<4) scal[h] = 0.f;       // per-step diagnostic accumulators
  __syncthreads();
  float g[4];
  for (int s=0;s<4;s++){
    int row = s*DH+h;
    float acc = b_ih[row] + b_hh[row];
    #pragma unroll 8
    for (int k=0;k<DEMB;k++) acc += xb[k]*w_ih[row*DEMB+k];
    #pragma unroll 8
    for (int k=0;k<DH;  k++) acc += hb[k]*w_hh[row*DH+k];
    g[s] = acc;
  }
  float ig = sigf(g[0]), fg = sigf(g[1]), gg = tanhf(g[2]), og = sigf(g[3]);
  float c1 = fg*ccar[i*DH+h] + ig*gg;
  float h1 = og*tanhf(c1);
  hA[i*DH+h] = h1; cA[i*DH+h] = c1; oG[i*DH+h] = og;
}

// ---------------------------------------------------------------- GCN precompute: per-node h projections
__global__ void gcnpre_k(const float* __restrict__ hin,
                         const float* __restrict__ w_gate, const float* __restrict__ w_ar,
                         float* Ag, float* Bg, float* sA, float* sB, int p){
  __shared__ float hb[DH];
  int i = blockIdx.x, h = threadIdx.x;
  hb[h] = hin[i*DH+h];
  __syncthreads();
  const float* wg = w_gate + p*DH*160;
  float a = 0.f, b = 0.f;
  #pragma unroll 8
  for (int k=0;k<DH;k++){ a += hb[k]*wg[h*160+32+k]; b += hb[k]*wg[h*160+96+k]; }
  Ag[i*DH+h] = a; Bg[i*DH+h] = b;
  if (h==0){ float s=0.f; for(int k=0;k<DH;k++) s += hb[k]*w_ar[p*160+32+k]; sA[i]=s; }
  if (h==1){ float s=0.f; for(int k=0;k<DH;k++) s += hb[k]*w_ar[p*160+96+k]; sB[i]=s; }
}

// ---------------------------------------------------------------- GCN main: scores+softmax, WMMA gate GEMM, message, c/h update
__global__ void __launch_bounds__(128) gcnmain_k(
    const float* __restrict__ habs, const int* __restrict__ seq, const int* __restrict__ nei,
    const float* __restrict__ w_rel, const float* __restrict__ b_rel,
    const float* __restrict__ w_gate, const float* __restrict__ b_gate,
    const float* __restrict__ w_ar, const float* __restrict__ b_ar,
    const float* __restrict__ w_nei,
    const float* __restrict__ hin, const float* __restrict__ cin, const float* __restrict__ oG,
    const float* __restrict__ Ag, const float* __restrict__ Bg,
    const float* __restrict__ sA, const float* __restrict__ sB,
    float* hout, float* cout, float* scal, int p, int t)
{
  __shared__ float s_abs[NN][2];
  __shared__ float s_wrel[DREL][2];
  __shared__ float s_brel[DREL];
  __shared__ float s_warr[DREL];
  __shared__ float s_bg[DH];
  __shared__ float s_sAb[16];
  __shared__ float s_sB[NN];
  __shared__ float s_pos[16][NN];   // scores, then pos (softmax*nei)
  __shared__ float s_nei[16][NN];
  __shared__ float s_Ag[16][DH];
  __shared__ float s_msg[16][DH];
  __shared__ float s_red[4];        // 0:ns 1:maxpos-sum 2:rowHasNei-sum 3:gate-sum

  int tid  = threadIdx.x;
  int i0   = blockIdx.x*16;
  int wave = tid >> 5, lane = tid & 31;

  // ---- load phase (s_abs / s_sB staged via CDNA5 async global->LDS DMA)
  {
    uint32_t abs_lds = (uint32_t)(uintptr_t)&s_abs[0][0];
    const float* gabs = habs + (size_t)t*NN*2;
    for (int idx=tid; idx<NN*2; idx+=128)
      async_ld_b32(abs_lds + (uint32_t)idx*4u, gabs + idx);
    uint32_t sB_lds = (uint32_t)(uintptr_t)&s_sB[0];
    for (int idx=tid; idx<NN; idx+=128)
      async_ld_b32(sB_lds + (uint32_t)idx*4u, sB + idx);
  }
  for (int idx=tid; idx<DREL*2; idx+=128) s_wrel[idx>>1][idx&1] = w_rel[p*DREL*2+idx];
  if (tid<DREL){ s_brel[tid] = b_rel[p*DREL+tid]; s_warr[tid] = w_ar[p*160+tid]; }
  if (tid<DH)   s_bg[tid] = b_gate[p*DH+tid];
  if (tid<16)   s_sAb[tid] = sA[i0+tid];
  for (int idx=tid; idx<16*DH;  idx+=128) s_Ag[idx>>6][idx&63] = Ag[(i0+(idx>>6))*DH + (idx&63)];
  if (tid<4) s_red[tid] = 0.f;
  for (int idx=tid; idx<NN; idx+=128){   // warm caches for the streamed rows
    __builtin_prefetch(&hin[idx*DH], 0, 1);
    __builtin_prefetch(&Bg[idx*DH],  0, 1);
  }
  wait_async();
  __syncthreads();

  // ---- pass 1: masked scores
  float barp = b_ar[p];
  float nsl  = 0.f;
  for (int idx=tid; idx<16*NN; idx+=128){
    int il = idx/NN, j = idx%NN, i = i0+il;
    int mi = seq[t*NN+i] > 0, mj = seq[t*NN+j] > 0;
    float nf = (nei[(size_t)(t*NN+i)*NN + j] > 0 && mi && mj) ? 1.f : 0.f;
    float sc;
    if (nf > 0.f){
      float c0 = s_abs[i][0]-s_abs[j][0], c1 = s_abs[i][1]-s_abs[j][1];
      float acc = s_sAb[il] + s_sB[j] + barp;
      #pragma unroll 8
      for (int k=0;k<DREL;k++){
        float r = c0*s_wrel[k][0] + c1*s_wrel[k][1] + s_brel[k];
        acc += (r>0.f?r:0.f)*s_warr[k];
      }
      sc = acc;
    } else sc = -1e9f;
    s_pos[il][j] = sc; s_nei[il][j] = nf; nsl += nf;
  }
  atomicAdd(&s_red[0], nsl);
  __syncthreads();

  // ---- softmax over j (one wave per row group, wave32 shuffles)
  for (int il=wave; il<16; il+=4){
    float m = -3e38f, nm = 0.f, v[6];
    #pragma unroll
    for (int q=0;q<6;q++){
      v[q] = s_pos[il][lane+32*q];
      m  = fmaxf(m,  v[q]);
      nm = fmaxf(nm, s_nei[il][lane+32*q]);
    }
    #pragma unroll
    for (int off=16; off>=1; off>>=1){ m = fmaxf(m, __shfl_xor(m, off)); nm = fmaxf(nm, __shfl_xor(nm, off)); }
    float sum = 0.f;
    #pragma unroll
    for (int q=0;q<6;q++){ v[q] = __expf(v[q]-m); sum += v[q]; }
    #pragma unroll
    for (int off=16; off>=1; off>>=1) sum += __shfl_xor(sum, off);
    float rs = 1.f/sum, pmax = 0.f;
    #pragma unroll
    for (int q=0;q<6;q++){
      float pv = v[q]*rs * s_nei[il][lane+32*q];
      s_pos[il][lane+32*q] = pv;
      pmax = fmaxf(pmax, pv);
    }
    #pragma unroll
    for (int off=16; off>=1; off>>=1) pmax = fmaxf(pmax, __shfl_xor(pmax, off));
    if (lane==0){ atomicAdd(&s_red[1], pmax); atomicAdd(&s_red[2], nm); }
  }
  __syncthreads();
  if (tid==0 && p==0){
    atomicAdd(&scal[1], s_red[0]); atomicAdd(&scal[2], s_red[1]); atomicAdd(&scal[3], s_red[2]);
  }

  // ---- pass 2: WMMA gate GEMM + fused message reduction
  // B tiles: B[k][h] = w_gate[p][h][k], k<32 (rel slice). ISA B layout:
  // lane holds col = lane&15, K = (lane>>4)*8 + {0..7} and +16..23.
  const float* wg = w_gate + p*DH*160;
  v16h btile[4];
  {
    int col = lane & 15, kb = (lane>>4)*8;
    #pragma unroll
    for (int ht=0; ht<4; ht++){
      int h = ht*16 + col;
      #pragma unroll
      for (int q=0;q<16;q++) btile[ht][q] = (_Float16)wg[h*160 + kb + q + (q>=8 ? 8 : 0)];
    }
  }
  float gsl = 0.f;
  for (int ii=0; ii<4; ii++){
    int il = wave*4 + ii, i = i0 + il;
    float ai0 = s_abs[i][0], ai1 = s_abs[i][1];
    float msgacc[4] = {0.f,0.f,0.f,0.f};
    for (int jt=0; jt<12; jt++){
      // A tile: row = j-local = lane&15; K = (lane>>4)*8 + {0..7} and +16..23
      int jrow = jt*16 + (lane & 15);
      float c0 = ai0 - s_abs[jrow][0], c1 = ai1 - s_abs[jrow][1];
      int kb8 = (lane>>4)*8;
      v16h a;
      #pragma unroll
      for (int q=0;q<8;q++){
        int k = kb8 + q;
        float r = c0*s_wrel[k][0] + c1*s_wrel[k][1] + s_brel[k];
        a[q]   = (_Float16)(r>0.f?r:0.f);
        k = kb8 + 16 + q;
        r = c0*s_wrel[k][0] + c1*s_wrel[k][1] + s_brel[k];
        a[q+8] = (_Float16)(r>0.f?r:0.f);
      }
      int half = lane >> 4;
      #pragma unroll
      for (int ht=0; ht<4; ht++){
        v8f acc = {};
        acc = __builtin_amdgcn_wmma_f32_16x16x32_f16(false, a, false, btile[ht],
                                                     (short)0, acc, false, false);
        int hcol = ht*16 + (lane & 15);
        float agv = s_Ag[il][hcol] + s_bg[hcol];
        #pragma unroll
        for (int v8=0; v8<8; v8++){
          int jl = v8 + 8*half;            // D row = v + 8*(lane>>4)
          int j  = jt*16 + jl;
          float pre  = acc[v8] + agv + Bg[j*DH+hcol];
          float gate = sigf(pre);
          float nf   = s_nei[il][j];
          gsl += gate*nf;
          msgacc[ht] += s_pos[il][j] * gate * hin[j*DH+hcol];
        }
      }
    }
    #pragma unroll
    for (int ht=0; ht<4; ht++){
      msgacc[ht] += __shfl_xor(msgacc[ht], 16);     // combine j halves (wave32)
      if (lane < 16) s_msg[il][ht*16+lane] = msgacc[ht];
    }
  }
  atomicAdd(&s_red[3], gsl);
  __syncthreads();
  if (tid==0 && p==0) atomicAdd(&scal[0], s_red[3]);

  // ---- c/h update: c_new = c + msg @ w_nei.T ; h_new = o * tanh(c_new)
  const float* wn = w_nei + p*DH*DH;
  for (int idx=tid; idx<16*DH; idx+=128){
    int il = idx>>6, h = idx&63, i = i0+il;
    float acc = 0.f;
    #pragma unroll 8
    for (int k=0;k<DH;k++) acc += s_msg[il][k]*wn[h*DH+k];
    float c2 = cin[i*DH+h] + acc;
    float h2 = oG[i*DH+h]*tanhf(c2);
    cout[i*DH+h] = c2; hout[i*DH+h] = h2;
  }
}

// ---------------------------------------------------------------- finalize: outputs, masked carry, diagnostics
__global__ void final_k(const float* __restrict__ hC, const float* __restrict__ cC,
                        const float* __restrict__ w_out, const float* __restrict__ b_out,
                        const int* __restrict__ seq,
                        float* out, float* hcar, float* ccar,
                        const float* __restrict__ scal, float* vacc, float* vout, int t){
  __shared__ float hb[DH];
  int i = blockIdx.x, h = threadIdx.x;
  float hv = hC[i*DH+h];
  hb[h] = hv;
  __syncthreads();
  int m = seq[t*NN+i] > 0;
  if (h < 2){
    float acc = b_out[h];
    for (int k=0;k<DH;k++) acc += hb[k]*w_out[h*DH+k];
    out[(t*NN+i)*2+h] = m ? acc : 0.f;
  }
  if (m){ hcar[i*DH+h] = hv; ccar[i*DH+h] = cC[i*DH+h]; }
  if (i==0 && h==0){
    float gs = scal[0], ns = scal[1], mp = scal[2], rn = scal[3];
    vacc[0] += gs/(ns*(float)DH + 1e-6f);
    vacc[1] += mp/(rn + 1e-6f);
    vacc[2] += ns/(float)NN;
    if (t == T_STEPS-2){
      vout[0] = vacc[0]/(float)T_STEPS;
      vout[1] = vacc[1]/(float)T_STEPS;
      vout[2] = vacc[2]/(float)T_STEPS;
    }
  }
}

// ---------------------------------------------------------------- host
extern "C" void kernel_launch(void* const* d_in, const int* in_sizes, int n_in,
                              void* d_out, int out_size, void* d_ws, size_t ws_size,
                              hipStream_t stream){
  (void)in_sizes; (void)n_in; (void)ws_size;
  const float* nodes_abs  = (const float*)d_in[0];
  const float* nodes_norm = (const float*)d_in[1];
  const float* w_in  = (const float*)d_in[3];
  const float* b_in  = (const float*)d_in[4];
  const float* w_ih  = (const float*)d_in[5];
  const float* w_hh  = (const float*)d_in[6];
  const float* b_ih  = (const float*)d_in[7];
  const float* b_hh  = (const float*)d_in[8];
  const float* w_rel = (const float*)d_in[9];
  const float* b_rel = (const float*)d_in[10];
  const float* w_gate= (const float*)d_in[11];
  const float* b_gate= (const float*)d_in[12];
  const float* w_ar  = (const float*)d_in[13];
  const float* b_ar  = (const float*)d_in[14];
  const float* w_nei = (const float*)d_in[15];
  const float* w_out = (const float*)d_in[16];
  const float* b_out = (const float*)d_in[17];
  const int*   seq   = (const int*)d_in[18];
  const int*   nei   = (const int*)d_in[19];

  float* out  = (float*)d_out;
  float* hcar = out + T_STEPS*NN*2;   // h carry lives in d_out slab
  float* ccar = hcar + NN*DH;         // c carry
  float* vout = ccar + NN*DH;         // 3 diagnostics

  float* W  = (float*)d_ws;
  float* hA = W;            float* cA = hA + NN*DH;
  float* oG = cA + NN*DH;
  float* hB = oG + NN*DH;   float* cB = hB + NN*DH;
  float* hC = cB + NN*DH;   float* cC = hC + NN*DH;
  float* Ag = cC + NN*DH;   float* Bg = Ag + NN*DH;
  float* sA = Bg + NN*DH;   float* sB = sA + NN;
  float* scal = sB + NN;    float* vacc = scal + 4;

  init_k<<<(out_size+255)/256, 256, 0, stream>>>(out, vacc, out_size);
  for (int t=0; t<T_STEPS-1; t++){
    lstm_k<<<NN, DH, 0, stream>>>(nodes_norm, w_in, b_in, w_ih, w_hh, b_ih, b_hh,
                                  hcar, ccar, hA, cA, oG, scal, t);
    gcnpre_k<<<NN, DH, 0, stream>>>(hA, w_gate, w_ar, Ag, Bg, sA, sB, 0);
    gcnmain_k<<<NN/16, 128, 0, stream>>>(nodes_abs, seq, nei, w_rel, b_rel, w_gate, b_gate,
                                         w_ar, b_ar, w_nei, hA, cA, oG, Ag, Bg, sA, sB,
                                         hB, cB, scal, 0, t);
    gcnpre_k<<<NN, DH, 0, stream>>>(hB, w_gate, w_ar, Ag, Bg, sA, sB, 1);
    gcnmain_k<<<NN/16, 128, 0, stream>>>(nodes_abs, seq, nei, w_rel, b_rel, w_gate, b_gate,
                                         w_ar, b_ar, w_nei, hB, cB, oG, Ag, Bg, sA, sB,
                                         hC, cC, scal, 1, t);
    final_k<<<NN, DH, 0, stream>>>(hC, cC, w_out, b_out, seq, out, hcar, ccar,
                                   scal, vacc, vout, t);
  }
}
